// SSIM_SMIC_58626303590530
// MI455X (gfx1250) — matrix operand order repaired and checked by reading, not compiled, not run on hardware
//
#include <hip/hip_runtime.h>
#include <hip/hip_bf16.h>

// CDNA5 / gfx1250 fused SSIM:
//   - TDM (tensor_load_to_lds) stages 26x32 f32 tiles global->LDS, double-
//     buffered across channels, paced with s_wait_tensorcnt.
//   - v_wmma_f32_16x16x32_f16 does both separable Gaussian passes as banded
//     GEMMs; LDS is the transposer between the passes.
//   - SSIM rational math runs per-element on identically laid-out fragments.
// One wave computes one 16x16 output tile across all 3 channels.

typedef _Float16 half16 __attribute__((ext_vector_type(16)));
typedef _Float16 half8  __attribute__((ext_vector_type(8)));
typedef float    f32x8  __attribute__((ext_vector_type(8)));
typedef unsigned int u32x4 __attribute__((ext_vector_type(4)));
typedef int      i32x4 __attribute__((ext_vector_type(4)));
typedef int      i32x8 __attribute__((ext_vector_type(8)));

#define WAVES_PER_BLOCK 4
#define TPB (WAVES_PER_BLOCK * 32)

// LDS row stride in f32 for TDM-staged tiles: 32 data + 1 pad DWORD
// (pad via D# pad_interval/pad_amount) -> conflict-free column gathers.
#define SROW 33

// Generic pointers into LDS carry the LDS byte offset in their low 32 bits
// (ISA 10.2, LDS aperture).
__device__ __forceinline__ unsigned lds_byte_of(const void* p) {
  return (unsigned)(unsigned long long)(uintptr_t)p;
}

// Issue one TDM 2D tile load: 32x26 f32 tile at gptr (tile start), row stride
// `stride` elements, remaining tensor extent rem_w x rem_h (hardware zero-fills
// out-of-bounds reads). LDS destination gets 1 DWORD of padding per 128B row.
__device__ __forceinline__ void tdm_load_tile_2d(
    unsigned lds_byte, const float* gptr,
    unsigned rem_w, unsigned rem_h, unsigned stride)
{
  unsigned long long ga = (unsigned long long)(uintptr_t)gptr;
  // Descriptor scalars must be wave-uniform SGPRs; they are per-wave uniform
  // by construction, make that explicit for codegen.
  unsigned ga_lo = (unsigned)__builtin_amdgcn_readfirstlane((int)(unsigned)ga);
  unsigned ga_hi = (unsigned)__builtin_amdgcn_readfirstlane((int)(unsigned)(ga >> 32));
  unsigned ldsb  = (unsigned)__builtin_amdgcn_readfirstlane((int)lds_byte);
  unsigned rw    = (unsigned)__builtin_amdgcn_readfirstlane((int)rem_w);
  unsigned rh    = (unsigned)__builtin_amdgcn_readfirstlane((int)rem_h);

  // D# group 0 (ISA 8.3): count=1 | lds_addr | global_addr[56:0] | type=2.
  u32x4 g0;
  g0.x = 1u;
  g0.y = ldsb;
  g0.z = ga_lo;
  g0.w = (ga_hi & 0x01ffffffu) | (2u << 30);

  // D# group 1 (ISA 8.4): data_size=4B (code 2), pad_enable, pad_interval
  // code 4 (=32 DWORDs), pad_amount code 0 (=1 DWORD); tensor_dim0/1 = rem;
  // tile_dim0=32, tile_dim1=26, tile_dim2=0; dim0 stride; dim1 stride unused.
  i32x8 g1;
  g1[0] = (int)((2u << 16) | (1u << 20) | (4u << 22));
  g1[1] = (int)((rw & 0xffffu) << 16);
  g1[2] = (int)(((rw >> 16) & 0xffffu) | ((rh & 0xffffu) << 16));
  g1[3] = (int)(((rh >> 16) & 0xffffu) | (32u << 16));
  g1[4] = (int)26u;
  g1[5] = (int)stride;
  g1[6] = 0;
  g1[7] = 0;

  i32x4 z4 = {0, 0, 0, 0};   // groups 2/3: zero (2D tensor)
#if defined(__clang_major__) && (__clang_major__ >= 23)
  i32x8 z8 = {};
  __builtin_amdgcn_tensor_load_to_lds(g0, g1, z4, z4, z8, 0);
#else
  __builtin_amdgcn_tensor_load_to_lds(g0, g1, z4, z4, 0);
#endif
}

__global__ __launch_bounds__(TPB) void ssim_wmma_kernel(
    const float* __restrict__ X, const float* __restrict__ Y,
    float* __restrict__ out)
{
  constexpr int H = 512, W = 512, OH = 502, OW = 502, CH = 3;
  constexpr float INV255 = 1.0f / 255.0f;
  // data scaled to [0,1] => C1=(K1*1)^2, C2=(K2*1)^2 (SSIM is scale invariant)
  constexpr float C1 = 1.0e-4f;
  constexpr float C2 = 9.0e-4f;

  // Per-wave private LDS. gX/gY: TDM-staged f32 tiles, double-buffered,
  // 32 rows x SROW stride (rows 26..31 pre-zeroed; TDM writes rows 0..25).
  // Vl: vertical-blur intermediates (row-major [row][32], cols 26..31 = 0).
  __shared__ __align__(16) float    gX[WAVES_PER_BLOCK][2][32 * SROW];
  __shared__ __align__(16) float    gY[WAVES_PER_BLOCK][2][32 * SROW];
  __shared__ __align__(32) _Float16 Vl[WAVES_PER_BLOCK][5][16 * 32];

  const int tid  = threadIdx.x;
  const int wave = tid >> 5;
  const int lane = tid & 31;
  const int sel  = lane >> 4;   // lane half (WMMA layout selector)
  const int mn   = lane & 15;   // M (A-frag) / N (B,C,D-frag) coordinate

  // Zero all LDS once: Vl padding cols and gX/gY rows 26..31 must be real
  // zeros (WMMA propagates NaN); TDM rewrites rows 0..25 every channel.
  for (int i = tid; i < WAVES_PER_BLOCK * 5 * 16 * 32; i += TPB)
    ((_Float16*)Vl)[i] = (_Float16)0.0f;
  for (int i = tid; i < WAVES_PER_BLOCK * 2 * 32 * SROW; i += TPB) {
    ((float*)gX)[i] = 0.0f;
    ((float*)gY)[i] = 0.0f;
  }

  // Tile assignment: 16 images x 32x32 tiles, 1 tile per wave.
  const int t  = blockIdx.x * WAVES_PER_BLOCK + wave;
  const int n  = t >> 10;
  const int ty = (t >> 5) & 31;
  const int tx = t & 31;
  const int r0 = ty * 16, c0 = tx * 16;

  // Gaussian window, matches _fspecial_gauss_1d(11, 1.5).
  float w[11];
  {
    float s = 0.f;
#pragma unroll
    for (int k = 0; k < 11; ++k) {
      float d = (float)(k - 5);
      w[k] = __expf(-d * d / 4.5f);
      s += w[k];
    }
#pragma unroll
    for (int k = 0; k < 11; ++k) w[k] /= s;
  }

  // Constant banded-Gaussian fragments (A: contiguous k order, B: ISA 7.12.2 table k order):
// A-frag (vertical): A[M][kk]=w[kk-M]; half i -> K = i + 16*sel, M = lane&15.
  // B-frag (horizontal): B[kk][c]=w[kk-c]; lane<16: halves->K {0..7,16..23},
  //   lane>=16: +8 on both ranges, N = lane&15.
  half16 Av, Bh;
#pragma unroll
  for (int i = 0; i < 16; ++i) {
    int kkA = i + 16 * sel;
    int dA  = kkA - mn;
    Av[i] = (_Float16)((dA >= 0 && dA <= 10) ? w[dA] : 0.0f);
    int kkB = ((i < 8) ? i : i + 8) + 8 * sel;
    int dB  = kkB - mn;
    Bh[i] = (_Float16)((dB >= 0 && dB <= 10) ? w[dB] : 0.0f);
  }

  const float* Xt = X + ((size_t)n * CH) * H * W + (size_t)r0 * W + c0;
  const float* Yt = Y + ((size_t)n * CH) * H * W + (size_t)r0 * W + c0;
  const unsigned remw = (unsigned)(W - c0);
  const unsigned remh = (unsigned)(H - r0);

  __syncthreads();  // LDS zeroing visible before any TDM writes / reads

  // Kick off channel-0 DMA.
  tdm_load_tile_2d(lds_byte_of(&gX[wave][0][0]), Xt, remw, remh, W);
  tdm_load_tile_2d(lds_byte_of(&gY[wave][0][0]), Yt, remw, remh, W);

  f32x8 acc = {};
  for (int ch = 0; ch < CH; ++ch) {
    const int buf = ch & 1;

    // Prefetch next channel into the other buffer, then wait for ours
    // (TENSORcnt decrements in order -> <=2 means channel ch's pair landed).
    if (ch + 1 < CH) {
      tdm_load_tile_2d(lds_byte_of(&gX[wave][buf ^ 1][0]),
                       Xt + (size_t)(ch + 1) * H * W, remw, remh, W);
      tdm_load_tile_2d(lds_byte_of(&gY[wave][buf ^ 1][0]),
                       Yt + (size_t)(ch + 1) * H * W, remw, remh, W);
      __builtin_amdgcn_s_wait_tensorcnt(2);
    } else {
      __builtin_amdgcn_s_wait_tensorcnt(0);
    }

    // ---- Vertical blur: V(16x26) = A_w(16x32) x S(32x16) per column group.
    // B-frag built directly from the staged f32 tile: per-lane column gather
    // (conflict-free thanks to the SROW=33 padded stride) + cvt to f16.
#pragma unroll
    for (int jg = 0; jg <= 16; jg += 16) {
      half16 bx, by;
#pragma unroll
      for (int k = 0; k < 16; ++k) {
        float xv = gX[wave][buf][(sel * 16 + k) * SROW + jg + mn];
        float yv = gY[wave][buf][(sel * 16 + k) * SROW + jg + mn];
        bx[k] = (_Float16)(xv * INV255);
        by[k] = (_Float16)(yv * INV255);
      }
#pragma unroll
      for (int s = 0; s < 5; ++s) {
        half16 bs;
        if      (s == 0) bs = bx;
        else if (s == 1) bs = by;
        else if (s == 2) bs = bx * bx;   // v_pk_mul_f16
        else if (s == 3) bs = by * by;
        else             bs = bx * by;
        f32x8 cz = {};
        f32x8 cv = __builtin_amdgcn_wmma_f32_16x16x32_f16(
            false, Av, false, bs, (short)0, cz, false, false);
        // C/D layout: element g -> row g+8*sel, col jg+mn. Keep cols < 26.
        if (jg == 0 || mn < 10) {
#pragma unroll
          for (int g = 0; g < 8; ++g)
            Vl[wave][s][(g + 8 * sel) * 32 + jg + mn] = (_Float16)cv[g];
        }
      }
    }
    __syncthreads();

    // ---- Horizontal blur: O(16x16) = V(16x32) x B_w(32x16), per signal.
    f32x8 O[5];
#pragma unroll
    for (int s = 0; s < 5; ++s) {
      const half8 alo = *(const half8*)&Vl[wave][s][mn * 32 + 8 * sel];
      const half8 ahi = *(const half8*)&Vl[wave][s][mn * 32 + 16 + 8 * sel];
      half16 a = __builtin_shufflevector(alo, ahi,
          0, 1, 2, 3, 4, 5, 6, 7, 8, 9, 10, 11, 12, 13, 14, 15);
      f32x8 cz = {};
      O[s] = __builtin_amdgcn_wmma_f32_16x16x32_f16(
          false, a, false, Bh, (short)0, cz, false, false);
    }

    // ---- Pointwise SSIM on identically-laid-out fragments; channel mean.
#pragma unroll
    for (int e = 0; e < 8; ++e) {
      float mu1 = O[0][e], mu2 = O[1][e];
      float m11 = mu1 * mu1, m22 = mu2 * mu2, m12 = mu1 * mu2;
      float s1  = O[2][e] - m11;
      float s2  = O[3][e] - m22;
      float s12 = O[4][e] - m12;
      float cs  = (2.0f * s12 + C2) / (s1 + s2 + C2);
      acc[e] += (2.0f * m12 + C1) / (m11 + m22 + C1) * cs;
    }
    __syncthreads();  // Vl reuse next channel
  }

  // ---- Store 1 - mean_c(ssim). C/D layout -> (row, col) in output.
#pragma unroll
  for (int e = 0; e < 8; ++e) {
    int row = r0 + e + 8 * sel;
    int col = c0 + mn;
    if (row < OH && col < OW)
      out[((size_t)n * OH + row) * OW + col] = 1.0f - acc[e] * (1.0f / 3.0f);
  }
}

extern "C" void kernel_launch(void* const* d_in, const int* in_sizes, int n_in,
                              void* d_out, int out_size, void* d_ws, size_t ws_size,
                              hipStream_t stream) {
  (void)in_sizes; (void)n_in; (void)d_ws; (void)ws_size; (void)out_size;
  const float* X = (const float*)d_in[0];
  const float* Y = (const float*)d_in[1];
  float* out = (float*)d_out;
  const int total_tiles = 16 * 32 * 32;            // images x 32x32 tiles
  dim3 grid(total_tiles / WAVES_PER_BLOCK);        // 4096 blocks
  ssim_wmma_kernel<<<grid, TPB, 0, stream>>>(X, Y, out);
}
